// LlamaDecoderLayer_15075335209655
// MI455X (gfx1250) — matrix operand order repaired and checked by reading, not compiled, not run on hardware
//
#include <hip/hip_runtime.h>
#include <hip/hip_bf16.h>

// ---------------------------------------------------------------------------
// Llama decoder layer for MI455X (gfx1250): bf16 WMMA + TDM async tensor loads
// B=2, S=2048, H=4096, NH=32, HD=128, I=11008
// ---------------------------------------------------------------------------

#define B_SZ 2
#define S_SZ 2048
#define H_SZ 4096
#define NH_SZ 32
#define HD_SZ 128
#define I_SZ 11008

typedef __attribute__((ext_vector_type(16))) __bf16 v16bf;
typedef __attribute__((ext_vector_type(8)))  float  v8f;
typedef __attribute__((ext_vector_type(4)))  unsigned u32x4;
typedef __attribute__((ext_vector_type(8)))  int      i32x8;
typedef __attribute__((ext_vector_type(4)))  int      i32x4;

// --- TDM availability (device pass only; host pass falls back cleanly) -----
#if defined(__has_builtin)
#if __has_builtin(__builtin_amdgcn_tensor_load_to_lds) && \
    __has_builtin(__builtin_amdgcn_s_wait_tensorcnt)
#define USE_TDM 1
#endif
#endif
#ifndef USE_TDM
#define USE_TDM 0
#endif

__device__ __forceinline__ void tdm_load_2d(u32x4 g0, i32x8 g1) {
#if USE_TDM
    i32x4 z4 = {0, 0, 0, 0};
#if __clang_major__ >= 23
    i32x8 z8 = {0, 0, 0, 0, 0, 0, 0, 0};
    __builtin_amdgcn_tensor_load_to_lds(g0, g1, z4, z4, z8, 0);
#else
    __builtin_amdgcn_tensor_load_to_lds(g0, g1, z4, z4, 0);
#endif
#else
    (void)g0; (void)g1;
#endif
}

__device__ __forceinline__ void tdm_wait() {
#if USE_TDM
    __builtin_amdgcn_s_wait_tensorcnt(0);
#endif
}

__device__ __forceinline__ v8f wmma_bf16(v16bf a, v16bf b, v8f c) {
    // D = A(16x32 bf16) * B(32x16 bf16) + C(16x16 f32)
    return __builtin_amdgcn_wmma_f32_16x16x32_bf16(
        /*neg_a=*/false, a, /*neg_b=*/false, b,
        /*c_mod=*/(short)0, c, /*reuse_a=*/false, /*reuse_b=*/false);
}

// Load a 16-element bf16 fragment from two 16-byte LDS/global chunks.
__device__ __forceinline__ v16bf ld_frag(const __bf16* p0, const __bf16* p1) {
    union { uint4 u[2]; v16bf v; } r;
    r.u[0] = *(const uint4*)p0;
    r.u[1] = *(const uint4*)p1;
    return r.v;
}

// ---------------------------------------------------------------------------
// RMSNorm: fp32 row in, bf16 row out.  One 256-thread block per row.
// ---------------------------------------------------------------------------
__global__ __launch_bounds__(256) void rmsnorm_kernel(
    const float* __restrict__ x, const float* __restrict__ w,
    __bf16* __restrict__ out, int H)
{
    __shared__ float red[256];
    const int row = blockIdx.x;
    const int tid = threadIdx.x;
    const float* xr = x + (size_t)row * H;
    float s = 0.f;
    for (int i = tid; i < H; i += 256) { float v = xr[i]; s += v * v; }
    red[tid] = s;
    __syncthreads();
    for (int o = 128; o > 0; o >>= 1) {
        if (tid < o) red[tid] += red[tid + o];
        __syncthreads();
    }
    const float rms = rsqrtf(red[0] / (float)H + 1e-6f);
    __bf16* orow = out + (size_t)row * H;
    for (int i = tid; i < H; i += 256)
        orow[i] = (__bf16)(xr[i] * rms * w[i]);
}

// ---------------------------------------------------------------------------
// RoPE (neox rotate-half) in place on bf16 [B,S,NH,HD].
// ---------------------------------------------------------------------------
__global__ __launch_bounds__(256) void rope_kernel(
    __bf16* __restrict__ x, const long long* __restrict__ positions)
{
    const int idx = blockIdx.x * 256 + threadIdx.x;   // B*S*NH*64 total
    const int i  = idx & 63;
    int t = idx >> 6;
    const int hh = t & (NH_SZ - 1);
    t >>= 5;
    const int s = t & (S_SZ - 1);
    const int b = t >> 11;
    const float pos = (float)positions[s];
    const float f = __expf(-__logf(10000.f) * (float)i * (1.f / 64.f));
    const float ang = pos * f;
    const float c = __cosf(ang), sn = __sinf(ang);
    const size_t base = (((size_t)(b * S_SZ + s)) * NH_SZ + hh) * HD_SZ;
    const float x1 = (float)x[base + i];
    const float x2 = (float)x[base + i + 64];
    x[base + i]      = (__bf16)(x1 * c - x2 * sn);
    x[base + i + 64] = (__bf16)(x2 * c + x1 * sn);
}

// ---------------------------------------------------------------------------
// GEMM: C[M,N] = A_bf16[M,K] @ W_fp32[N,K]^T
// Block 256 threads (8 waves), tile 128(M) x 128(N), K step 32.
// Wave grid 4(M) x 2(N): each wave 32x64 (2x4 WMMA tiles).
// A tile staged by the Tensor Data Mover (LDS row padding via pad fields);
// W tile converted fp32->bf16 by the VALU staging path + prefetch.
// MODE 0: bf16 out; MODE 1: fp32 = gemm + res; MODE 2: bf16 = silu(aux)*gemm.
// ---------------------------------------------------------------------------
template <int MODE>
__global__ __launch_bounds__(256) void gemm_bf16_kernel(
    const __bf16* __restrict__ A, const float* __restrict__ W,
    int M, int N, int K,
    const float* __restrict__ res, const __bf16* __restrict__ aux,
    float* __restrict__ outF, __bf16* __restrict__ outB)
{
    __shared__ __bf16 As[128 * 40];   // 128 rows (M), 32 K + 8 pad
    __shared__ __bf16 Bs[128 * 40];   // 128 rows (N), 32 K + 8 pad

    const int tid  = threadIdx.x;
    const int lane = tid & 31;
    const int wid  = tid >> 5;
    const int wm   = wid & 3;         // 0..3 -> 32-row slice
    const int wn   = wid >> 2;        // 0..1 -> 64-col slice
    const int m0   = blockIdx.y * 128;
    const int n0   = blockIdx.x * 128;
    const int mr   = lane & 15;
    const int ka   = (lane < 16) ? 0 : 8;    // A frag: K base
    const int kb   = (lane < 16) ? 0 : 8;    // B frag: K base

    v8f c[2][4] = {};

#if USE_TDM && defined(__HIP_DEVICE_COMPILE__)
    // TDM descriptor group 1 (tile geometry), built once.
    // data_size=2B, pad_enable, pad_interval=16 DW (64B row), pad_amount=4 DW.
    const unsigned as_lds = (unsigned)(uintptr_t)(void*)As;
    i32x8 g1a;
    g1a[0] = (1 << 16) | (1 << 20) | (3 << 22) | (3 << 25);
    g1a[1] = (int)(((unsigned)K & 0xFFFFu) << 16);                 // tdim0 lo
    g1a[2] = (int)(((unsigned)K >> 16) | (((unsigned)M & 0xFFFFu) << 16));
    g1a[3] = (int)((((unsigned)M >> 16) & 0xFFFFu) | (32u << 16)); // tile_dim0
    g1a[4] = 128;                                                  // tile_dim1
    g1a[5] = K;                                                    // stride lo
    g1a[6] = 0;
    g1a[7] = 0;
#endif

    for (int k0 = 0; k0 < K; k0 += 32) {
#if USE_TDM && defined(__HIP_DEVICE_COMPILE__)
        // --- stage A (bf16 128x32 -> padded LDS) via Tensor Data Mover
        if (wid == 0) {
            const unsigned long long ga =
                (unsigned long long)(uintptr_t)(A + (size_t)m0 * K + k0);
            u32x4 g0;
            g0.x = 1u;                                  // count=1 (valid)
            g0.y = as_lds;                              // LDS byte address
            g0.z = (unsigned)ga;                        // global addr lo
            g0.w = (unsigned)(ga >> 32) | (2u << 30);   // addr hi | type=2
            tdm_load_2d(g0, g1a);
        }
#else
        // --- stage A cooperatively: thread -> row tid/2, 16 cols
        {
            const int r = tid >> 1, cg = (tid & 1) * 16;
            const uint4* apg = (const uint4*)(A + (size_t)(m0 + r) * K + k0 + cg);
            *(uint4*)(&As[r * 40 + cg])     = apg[0];
            *(uint4*)(&As[r * 40 + cg + 8]) = apg[1];
        }
#endif
        // --- stage W (fp32 -> bf16, 128x32): thread -> row tid/2, 16 cols
        {
            const int r = tid >> 1, cg = (tid & 1) * 16;
            const float4* wp = (const float4*)(W + (size_t)(n0 + r) * K + k0 + cg);
            float4 f0 = wp[0], f1 = wp[1], f2 = wp[2], f3 = wp[3];
            if (k0 + 32 < K) __builtin_prefetch(wp + 8, 0, 1);  // next K slab
            __bf16 tmp[16];
            tmp[0]=(__bf16)f0.x; tmp[1]=(__bf16)f0.y; tmp[2]=(__bf16)f0.z; tmp[3]=(__bf16)f0.w;
            tmp[4]=(__bf16)f1.x; tmp[5]=(__bf16)f1.y; tmp[6]=(__bf16)f1.z; tmp[7]=(__bf16)f1.w;
            tmp[8]=(__bf16)f2.x; tmp[9]=(__bf16)f2.y; tmp[10]=(__bf16)f2.z; tmp[11]=(__bf16)f2.w;
            tmp[12]=(__bf16)f3.x; tmp[13]=(__bf16)f3.y; tmp[14]=(__bf16)f3.z; tmp[15]=(__bf16)f3.w;
            *(uint4*)(&Bs[r * 40 + cg])     = *(const uint4*)(&tmp[0]);
            *(uint4*)(&Bs[r * 40 + cg + 8]) = *(const uint4*)(&tmp[8]);
        }
#if USE_TDM && defined(__HIP_DEVICE_COMPILE__)
        if (wid == 0) tdm_wait();
#endif
        __syncthreads();

        v16bf a[2];
#pragma unroll
        for (int im = 0; im < 2; ++im) {
            const __bf16* ap = &As[(wm * 32 + im * 16 + mr) * 40];
            a[im] = ld_frag(ap + ka, ap + ka + 16);
        }
#pragma unroll
        for (int in = 0; in < 4; ++in) {
            const __bf16* bp = &Bs[(wn * 64 + in * 16 + mr) * 40];
            const v16bf b = ld_frag(bp + kb, bp + kb + 16);
#pragma unroll
            for (int im = 0; im < 2; ++im)
                c[im][in] = wmma_bf16(a[im], b, c[im][in]);
        }
        __syncthreads();
    }

    // --- epilogue: c-layout VGPR i -> M = i (lanes 0-15) / 8+i (lanes 16-31)
    const int mbase = ((lane < 16) ? 0 : 8);
#pragma unroll
    for (int im = 0; im < 2; ++im) {
#pragma unroll
        for (int in = 0; in < 4; ++in) {
            const int mb = m0 + wm * 32 + im * 16 + mbase;
            const int nb = n0 + wn * 64 + in * 16 + mr;
#pragma unroll
            for (int i = 0; i < 8; ++i) {
                const size_t off = (size_t)(mb + i) * N + nb;
                const float v = c[im][in][i];
                if (MODE == 0) {
                    outB[off] = (__bf16)v;
                } else if (MODE == 1) {
                    outF[off] = v + res[off];
                } else {
                    const float g = (float)aux[off];
                    const float sg = g / (1.f + __expf(-g));
                    outB[off] = (__bf16)(sg * v);
                }
            }
        }
    }
}

// ---------------------------------------------------------------------------
// Flash attention (causal) with WMMA.  Block: 128 threads (4 waves) handles
// (b, h, 64 q rows); each wave owns 16 q rows.  Key blocks of 32.
// K tile staged by the TDM (padded rows); V transposed by the VALU path.
// Q,K,V,O: bf16 [B,S,NH,HD].
// ---------------------------------------------------------------------------
__global__ __launch_bounds__(128) void attn_kernel(
    const __bf16* __restrict__ Q, const __bf16* __restrict__ Kb,
    const __bf16* __restrict__ Vb, __bf16* __restrict__ O)
{
    __shared__ __bf16 Kt[32 * 136];     // [key][hd], padded
    __shared__ __bf16 Vt[128 * 40];     // [hd][key], padded (transposed V)
    __shared__ float  Sm[4][16 * 32];   // per-wave score tile
    __shared__ __bf16 Pm[4][16 * 40];   // per-wave prob tile (bf16, padded)
    __shared__ float  alpha_s[4][16];
    __shared__ float  linv_s[4][16];

    const int tid  = threadIdx.x;
    const int lane = tid & 31;
    const int wid  = tid >> 5;
    const int h    = blockIdx.y;
    const int b    = blockIdx.z;
    const int q0   = blockIdx.x * 64 + wid * 16;
    const int mr   = lane & 15;
    const int ka   = (lane < 16) ? 0 : 8;
    const int kb   = (lane < 16) ? 0 : 8;
    const size_t rs = (size_t)NH_SZ * HD_SZ;   // seq stride (4096)

    // --- load Q fragments once (4 chunks of K=32 over HD=128)
    v16bf qf[4];
    {
        const __bf16* qr = Q + ((size_t)(b * S_SZ + q0 + mr)) * rs + (size_t)h * HD_SZ;
#pragma unroll
        for (int cch = 0; cch < 4; ++cch)
            qf[cch] = ld_frag(qr + cch * 32 + ka, qr + cch * 32 + ka + 16);
    }

    v8f o[8];
#pragma unroll
    for (int i = 0; i < 8; ++i) o[i] = (v8f){};
    float mrow = -3.0e38f, lrow = 0.f;
    const float scale = 0.08838834764831845f;  // 1/sqrt(128)
    const int mbase = ((lane < 16) ? 0 : 8);
    const int kend = (blockIdx.x + 1) * 64;    // causal extent for this block

#if USE_TDM && defined(__HIP_DEVICE_COMPILE__)
    // K-tile descriptor: 32 rows x 128 bf16, row stride 4096 elements,
    // LDS rows padded 256B -> 272B (pad_interval=64 DW, pad_amount=4 DW).
    const unsigned kt_lds = (unsigned)(uintptr_t)(void*)Kt;
    i32x8 g1k;
    g1k[0] = (1 << 16) | (1 << 20) | (5 << 22) | (3 << 25);
    g1k[1] = (int)(128u << 16);   // tensor_dim0 = 128 (lo16)
    g1k[2] = (int)(32u << 16);    // tensor_dim0 hi | tensor_dim1 = 32
    g1k[3] = (int)(128u << 16);   // tensor_dim1 hi | tile_dim0 = 128
    g1k[4] = 32;                  // tile_dim1 = 32, tile_dim2 = 0
    g1k[5] = 4096;                // tensor_dim0_stride lo
    g1k[6] = 0;
    g1k[7] = 0;
#endif

    for (int k0 = 0; k0 < kend; k0 += 32) {
#if USE_TDM && defined(__HIP_DEVICE_COMPILE__)
        // --- stage K tile via Tensor Data Mover (wave 0 issues)
        if (wid == 0) {
            const unsigned long long ga = (unsigned long long)(uintptr_t)
                (Kb + ((size_t)(b * S_SZ + k0)) * rs + (size_t)h * HD_SZ);
            u32x4 g0;
            g0.x = 1u;
            g0.y = kt_lds;
            g0.z = (unsigned)ga;
            g0.w = (unsigned)(ga >> 32) | (2u << 30);
            tdm_load_2d(g0, g1k);
        }
#else
        // --- stage K tile [32 keys][128 hd] bf16 (straight copy)
        {
            const int key = tid >> 2, g = tid & 3;
            const __bf16* kp = Kb + ((size_t)(b * S_SZ + k0 + key)) * rs
                                  + (size_t)h * HD_SZ + g * 32;
            const uint4* ks = (const uint4*)kp;
            uint4* kd = (uint4*)(&Kt[key * 136 + g * 32]);
            kd[0] = ks[0]; kd[1] = ks[1]; kd[2] = ks[2]; kd[3] = ks[3];
        }
#endif
        // --- stage V transposed: Vt[hd][key]
        {
            const int j = tid & 31, half = tid >> 5;
            const __bf16* vp = Vb + ((size_t)(b * S_SZ + k0 + j)) * rs
                                  + (size_t)h * HD_SZ + half * 32;
#pragma unroll
            for (int n = 0; n < 32; ++n)
                Vt[(half * 32 + n) * 40 + j] = vp[n];
        }
#if USE_TDM && defined(__HIP_DEVICE_COMPILE__)
        if (wid == 0) tdm_wait();
#endif
        __syncthreads();

        // --- scores: S(16x32) = Q(16x128) @ K^T(128x32), scaled + masked
#pragma unroll
        for (int nt = 0; nt < 2; ++nt) {
            v8f sc = {};
#pragma unroll
            for (int cch = 0; cch < 4; ++cch) {
                const __bf16* bp = &Kt[(nt * 16 + mr) * 136 + cch * 32 + kb];
                const v16bf bfr = ld_frag(bp, bp + 16);
                sc = wmma_bf16(qf[cch], bfr, sc);
            }
#pragma unroll
            for (int i = 0; i < 8; ++i) {
                const int qrow = q0 + mbase + i;
                const int kcol = k0 + nt * 16 + mr;
                float v = sc[i] * scale;
                if (kcol > qrow) v = -3.0e38f;
                Sm[wid][(mbase + i) * 32 + nt * 16 + mr] = v;
            }
        }
        __syncthreads();

        // --- online softmax: lanes 0-15 each own one q row
        if (lane < 16) {
            const float* srow = &Sm[wid][lane * 32];
            float tmax = -3.0e38f;
#pragma unroll
            for (int j = 0; j < 32; ++j) tmax = fmaxf(tmax, srow[j]);
            const float mnew = fmaxf(mrow, tmax);
            const float a = __expf(mrow - mnew);
            float ssum = 0.f;
#pragma unroll
            for (int j = 0; j < 32; ++j) {
                const float e = __expf(srow[j] - mnew);
                ssum += e;
                Pm[wid][lane * 40 + j] = (__bf16)e;
            }
            lrow = lrow * a + ssum;
            mrow = mnew;
            alpha_s[wid][lane] = a;
        }
        __syncthreads();

        // --- rescale running output by alpha[row]
#pragma unroll
        for (int i = 0; i < 8; ++i) {
            const float a = alpha_s[wid][mbase + i];
#pragma unroll
            for (int nt = 0; nt < 8; ++nt) o[nt][i] *= a;
        }

        // --- O += P(16x32) @ V(32x128)
        {
            const __bf16* pp = &Pm[wid][mr * 40 + ka];
            const v16bf pa = ld_frag(pp, pp + 16);
#pragma unroll
            for (int nt = 0; nt < 8; ++nt) {
                const __bf16* vp2 = &Vt[(nt * 16 + mr) * 40 + kb];
                const v16bf vfr = ld_frag(vp2, vp2 + 16);
                o[nt] = wmma_bf16(pa, vfr, o[nt]);
            }
        }
        __syncthreads();   // protect Kt/Vt/Sm/Pm before next stage
    }

    if (lane < 16) linv_s[wid][lane] = 1.0f / lrow;
    __syncthreads();

#pragma unroll
    for (int i = 0; i < 8; ++i) {
        const int rl = mbase + i;
        const float li = linv_s[wid][rl];
#pragma unroll
        for (int nt = 0; nt < 8; ++nt) {
            const size_t off = ((size_t)(b * S_SZ + q0 + rl)) * rs
                             + (size_t)h * HD_SZ + nt * 16 + mr;
            O[off] = (__bf16)(o[nt][i] * li);
        }
    }
}

// ---------------------------------------------------------------------------
// Host-side launcher
// ---------------------------------------------------------------------------
extern "C" void kernel_launch(void* const* d_in, const int* in_sizes, int n_in,
                              void* d_out, int out_size, void* d_ws, size_t ws_size,
                              hipStream_t stream) {
    (void)in_sizes; (void)n_in; (void)out_size; (void)ws_size;

    const float*      hs   = (const float*)d_in[0];
    const long long*  pos  = (const long long*)d_in[1];
    const float*      wq   = (const float*)d_in[2];
    const float*      wk   = (const float*)d_in[3];
    const float*      wv   = (const float*)d_in[4];
    const float*      wo   = (const float*)d_in[5];
    const float*      wg   = (const float*)d_in[6];
    const float*      wu   = (const float*)d_in[7];
    const float*      wd   = (const float*)d_in[8];
    const float*      ln1  = (const float*)d_in[9];
    const float*      ln2  = (const float*)d_in[10];
    float*            outp = (float*)d_out;

    const int M = B_SZ * S_SZ;                        // 4096 token rows
    const size_t szHbf = (size_t)M * H_SZ * sizeof(__bf16);
    const size_t szHf  = (size_t)M * H_SZ * sizeof(float);
    const size_t szIbf = (size_t)M * I_SZ * sizeof(__bf16);

    char* w = (char*)d_ws;
    __bf16* xn   = (__bf16*)w;  w += szHbf;           // rmsnorm out (reused)
    __bf16* qb   = (__bf16*)w;  w += szHbf;
    __bf16* kbuf = (__bf16*)w;  w += szHbf;
    __bf16* vbuf = (__bf16*)w;  w += szHbf;
    __bf16* attn = (__bf16*)w;  w += szHbf;
    float*  xres = (float*)w;   w += szHf;            // residual after attention
    __bf16* gbuf = (__bf16*)w;  w += szIbf;
    __bf16* hbuf = (__bf16*)w;  w += szIbf;

    const dim3 blk256(256), blk128(128);
    const dim3 gH(H_SZ / 128, M / 128);               // GEMM grids (N, M)
    const dim3 gI(I_SZ / 128, M / 128);
    const int ropeBlocks = (B_SZ * S_SZ * NH_SZ * 64) / 256;

    // 1) rmsnorm(hidden) -> xn (bf16)
    rmsnorm_kernel<<<M, blk256, 0, stream>>>(hs, ln1, xn, H_SZ);

    // 2) q,k,v = xn @ w{q,k,v}^T  (bf16 out)
    gemm_bf16_kernel<0><<<gH, blk256, 0, stream>>>(xn, wq, M, H_SZ, H_SZ, nullptr, nullptr, nullptr, qb);
    gemm_bf16_kernel<0><<<gH, blk256, 0, stream>>>(xn, wk, M, H_SZ, H_SZ, nullptr, nullptr, nullptr, kbuf);
    gemm_bf16_kernel<0><<<gH, blk256, 0, stream>>>(xn, wv, M, H_SZ, H_SZ, nullptr, nullptr, nullptr, vbuf);

    // 3) RoPE on q and k (in place)
    rope_kernel<<<ropeBlocks, blk256, 0, stream>>>(qb, pos);
    rope_kernel<<<ropeBlocks, blk256, 0, stream>>>(kbuf, pos);

    // 4) causal flash attention -> attn (bf16)
    attn_kernel<<<dim3(S_SZ / 64, NH_SZ, B_SZ), blk128, 0, stream>>>(qb, kbuf, vbuf, attn);

    // 5) x = attn @ wo^T + hidden   (fp32, fused residual)
    gemm_bf16_kernel<1><<<gH, blk256, 0, stream>>>(attn, wo, M, H_SZ, H_SZ, hs, nullptr, xres, nullptr);

    // 6) y = rmsnorm2(x) -> xn (bf16, buffer reuse)
    rmsnorm_kernel<<<M, blk256, 0, stream>>>(xres, ln2, xn, H_SZ);

    // 7) g = y @ w_gate^T (bf16)
    gemm_bf16_kernel<0><<<gI, blk256, 0, stream>>>(xn, wg, M, I_SZ, H_SZ, nullptr, nullptr, nullptr, gbuf);

    // 8) h = silu(g) * (y @ w_up^T)  (fused epilogue, bf16)
    gemm_bf16_kernel<2><<<gI, blk256, 0, stream>>>(xn, wu, M, I_SZ, H_SZ, nullptr, gbuf, nullptr, hbuf);

    // 9) out = h @ w_down^T + x  (fp32, fused residual)
    gemm_bf16_kernel<1><<<gH, blk256, 0, stream>>>(hbuf, wd, M, H_SZ, I_SZ, xres, nullptr, outp, nullptr);
}
